// NTXentLoss_38113539784917
// MI455X (gfx1250) — matrix operand order repaired and checked by reading, not compiled, not run on hardware
//
#include <hip/hip_runtime.h>

typedef __attribute__((ext_vector_type(4)))  _Float16 v4h;
typedef __attribute__((ext_vector_type(8)))  _Float16 v8h;
typedef __attribute__((ext_vector_type(16))) _Float16 v16h;
typedef __attribute__((ext_vector_type(8)))  float    v8f;

#define DDIM 128          // feature dim (fixed by reference)
#define NSPLIT 8          // column splits per row tile; (N/16)/NSPLIT must be even

// ---------------------------------------------------------------------------
// Kernel 1: row-normalize z=[z_i; z_j] -> f16 zn, zero accumulators & output.
// ---------------------------------------------------------------------------
__global__ __launch_bounds__(32)
void ntxent_normalize(const float* __restrict__ zi,
                      const float* __restrict__ zj,
                      _Float16* __restrict__ zn16,
                      float* __restrict__ rowsum,
                      float* __restrict__ out,
                      int B) {
  const int r    = blockIdx.x;      // row 0..N-1
  const int lane = threadIdx.x;     // 0..31

  const float* src = (r < B) ? (zi + (size_t)r * DDIM)
                             : (zj + (size_t)(r - B) * DDIM);
  float4 v = ((const float4*)src)[lane];
  float ss = v.x * v.x + v.y * v.y + v.z * v.z + v.w * v.w;
  #pragma unroll
  for (int m = 16; m > 0; m >>= 1) ss += __shfl_xor(ss, m, 32);

  float norm = fmaxf(sqrtf(ss), 1e-8f);   // F.cosine_similarity eps
  float inv  = 1.0f / norm;

  v4h h;
  h[0] = (_Float16)(v.x * inv);
  h[1] = (_Float16)(v.y * inv);
  h[2] = (_Float16)(v.z * inv);
  h[3] = (_Float16)(v.w * inv);
  ((v4h*)(zn16 + (size_t)r * DDIM))[lane] = h;

  if (lane == 0) rowsum[r] = 0.0f;
  if (r == 0 && lane == 0) out[0] = 0.0f;
}

// ---------------------------------------------------------------------------
// Helpers for the fused GEMM kernel. All indices compile-time constant so
// fragments live purely in VGPRs (no scratch).
// ---------------------------------------------------------------------------
__device__ __forceinline__
void load_btile(v16h (&Bf)[4], const _Float16* __restrict__ zn16,
                int colBase, int g, int l) {
  const _Float16* brow = zn16 + (size_t)(colBase + l) * DDIM;
  #pragma unroll
  for (int kc = 0; kc < 4; ++kc) {
    v8h lo = *(const v8h*)(brow + kc * 32 + g * 8);       // K {0..7 | 8..15}
    v8h hi = *(const v8h*)(brow + kc * 32 + 16 + g * 8);  // K {16..23 | 24..31}
    #pragma unroll
    for (int i = 0; i < 8; ++i) { Bf[kc][i] = lo[i]; Bf[kc][i + 8] = hi[i]; }
  }
}

__device__ __forceinline__
void compute_tile(const v16h (&A)[4], const v16h (&Bf)[4], float (&s)[8]) {
  v8f c = {};
  #pragma unroll
  for (int kc = 0; kc < 4; ++kc) {
    c = __builtin_amdgcn_wmma_f32_16x16x32_f16(
            false, A[kc], false, Bf[kc], (short)0, c, false, false);
  }
  // s += exp(sim/T - 2);  sim/T = 2*c <= 2, so the exp arg is <= 0.
  #pragma unroll
  for (int v = 0; v < 8; ++v)
    s[v] += __expf(fmaf(2.0f, c[v], -2.0f));
}

// ---------------------------------------------------------------------------
// Kernel 2: fused sim-tile GEMM (WMMA f16->f32) + exp-sum row reduction.
// grid.x = N/16 row tiles, grid.y = NSPLIT column splits, 1 wave per block.
// Software-pipelined x2: B-tile loads for the next tile issue before the
// WMMAs of the current tile, hiding L2 latency. Diagonal & positive terms
// are corrected in the finalize kernel (no special cases in the hot loop).
// ---------------------------------------------------------------------------
__global__ __launch_bounds__(32)
void ntxent_sim(const _Float16* __restrict__ zn16,
                float* __restrict__ rowsum,
                int colTilesPerSplit) {
  const int lane = threadIdx.x;
  const int g    = lane >> 4;     // half-wave group
  const int l    = lane & 15;
  const int rowBase = blockIdx.x * 16;

  // A fragments (16-bit A 16x32 lane layout, ISA 7.12.2): loaded once.
  const _Float16* arow = zn16 + (size_t)(rowBase + l) * DDIM;
  v16h A[4];
  #pragma unroll
  for (int kc = 0; kc < 4; ++kc) {
    v8h lo = *(const v8h*)(arow + kc * 32 + g * 8);        // K {0..7 | 8..15}
    v8h hi = *(const v8h*)(arow + kc * 32 + 16 + g * 8);   // K {16..23 | 24..31}
    #pragma unroll
    for (int i = 0; i < 8; ++i) { A[kc][i] = lo[i]; A[kc][i + 8] = hi[i]; }
  }

  float s[8];
  #pragma unroll
  for (int v = 0; v < 8; ++v) s[v] = 0.0f;

  const int ct0  = blockIdx.y * colTilesPerSplit;
  const int last = colTilesPerSplit - 1;

  v16h B0[4], B1[4];               // named buffers -> constant indexing
  load_btile(B0, zn16, ct0 * 16, g, l);

  for (int ct = 0; ct < colTilesPerSplit; ct += 2) {   // count is even
    load_btile(B1, zn16, (ct0 + ct + 1) * 16, g, l);   // prefetch odd tile
    compute_tile(A, B0, s);                            // even tile

    const int nx = (ct + 2 <= last) ? (ct + 2) : last; // clamped prefetch
    load_btile(B0, zn16, (ct0 + nx) * 16, g, l);
    compute_tile(A, B1, s);                            // odd tile
  }

  // Reduce each row's partial sum across its 16-lane group, then atomic-add.
  #pragma unroll
  for (int v = 0; v < 8; ++v) {
    float t = s[v];
    t += __shfl_xor(t, 1, 32);
    t += __shfl_xor(t, 2, 32);
    t += __shfl_xor(t, 4, 32);
    t += __shfl_xor(t, 8, 32);
    if (l == 0) atomicAdd(&rowsum[rowBase + v + 8 * g], t);
  }
}

// ---------------------------------------------------------------------------
// Kernel 3: one wave per row.
//   diag = 2*<zn_r, zn_r>, pos = 2*<zn_r, zn_partner>   (f32 dots of f16 data)
//   loss_r = log(rowsum_r - exp(diag-2)) + 2 - pos ;  out = mean_r loss_r
// ---------------------------------------------------------------------------
__global__ __launch_bounds__(32)
void ntxent_finalize(const _Float16* __restrict__ zn16,
                     const float* __restrict__ rowsum,
                     float* __restrict__ out, int B) {
  const int r    = blockIdx.x;
  const int lane = threadIdx.x;
  const int N    = 2 * B;
  const int partner = (r < B) ? (r + B) : (r - B);

  v4h ha = ((const v4h*)(zn16 + (size_t)r * DDIM))[lane];
  v4h hb = ((const v4h*)(zn16 + (size_t)partner * DDIM))[lane];
  float ds = 0.0f, dp = 0.0f;
  #pragma unroll
  for (int i = 0; i < 4; ++i) {
    float fa = (float)ha[i], fb = (float)hb[i];
    ds = fmaf(fa, fa, ds);
    dp = fmaf(fa, fb, dp);
  }
  #pragma unroll
  for (int m = 16; m > 0; m >>= 1) {
    ds += __shfl_xor(ds, m, 32);
    dp += __shfl_xor(dp, m, 32);
  }

  if (lane == 0) {
    float sum  = rowsum[r] - __expf(fmaf(2.0f, ds, -2.0f));  // drop diagonal
    float loss = (logf(sum) + 2.0f) - 2.0f * dp;
    atomicAdd(out, loss * (1.0f / (float)N));
  }
}

// ---------------------------------------------------------------------------
extern "C" void kernel_launch(void* const* d_in, const int* in_sizes, int n_in,
                              void* d_out, int out_size, void* d_ws, size_t ws_size,
                              hipStream_t stream) {
  const float* zi = (const float*)d_in[0];
  const float* zj = (const float*)d_in[1];
  float* out = (float*)d_out;

  const int B = in_sizes[0] / DDIM;   // 4096
  const int N = 2 * B;                // 8192

  // Workspace: zn16 (N*128 halves = 2 MB), rowsum (N floats)
  _Float16* zn16   = (_Float16*)d_ws;
  float*    rowsum = (float*)((char*)d_ws + (size_t)N * DDIM * sizeof(_Float16));

  ntxent_normalize<<<N, 32, 0, stream>>>(zi, zj, zn16, rowsum, out, B);

  const int rowTiles = N / 16;                       // 512
  const int colTilesPerSplit = rowTiles / NSPLIT;    // 64 (even)
  dim3 grid(rowTiles, NSPLIT);
  ntxent_sim<<<grid, 32, 0, stream>>>(zn16, rowsum, colTilesPerSplit);

  ntxent_finalize<<<N, 32, 0, stream>>>(zn16, rowsum, out, B);
}
